// pre_GCNModel_edge_3eosZ_32504312496843
// MI455X (gfx1250) — matrix operand order repaired and checked by reading, not compiled, not run on hardware
//
#include <hip/hip_runtime.h>
#include <math.h>

typedef __attribute__((ext_vector_type(16))) __bf16 v16bf;
typedef __attribute__((ext_vector_type(8)))  float  v8f;

static constexpr int NN   = 10000;
static constexpr int EE   = 160000;
static constexpr int GG   = 100;
static constexpr int NPG  = NN / GG;   // 100
static constexpr int EPG  = EE / GG;   // 1600

// conv1: CIN=16, KF=10 -> KPAD=160, 5 K-blocks ; conv2: CIN=32, KF=9 -> KPAD=288, 9 K-blocks
static constexpr int KBLK1 = 5;
static constexpr int KBLK2 = 9;
static constexpr int FRAG_ELEMS = 32 * 16;              // per (kb,ntile): 32 lanes x 16 bf16
static constexpr int NSW1 = KBLK1 * 2 * FRAG_ELEMS;     // 5120 bf16
static constexpr int NSW2 = KBLK2 * 2 * FRAG_ELEMS;     // 9216 bf16

__device__ __forceinline__ float leaky(float v) { return v >= 0.0f ? v : 0.05f * v; }

// ---------------- zero accumulators ----------------
__global__ void k_init(float* __restrict__ a, float* __restrict__ b, int n) {
  int i = blockIdx.x * blockDim.x + threadIdx.x;
  int stride = gridDim.x * blockDim.x;
  for (; i < n; i += stride) { a[i] = 0.0f; b[i] = 0.0f; }
}

// ---------------- build bf16 weights PRE-SWIZZLED into WMMA B-fragment order ----------------
// Fragment layout: t = ((kb*2 + nt)*32 + lane)*16 + j
//   K = kb*32 + (lane>=16 ? 16 : 0) + j   (contiguous k order, same as A: lane<16 -> K 0..15, lane>=16 -> K 16..31)
//   N = nt*16 + (lane & 15)
// Logical weight row r=K: conv1 f=r/16,i=r%16 (f<8: nn_w[f,i*32+N], f==8: nn_b, f==9: pad 0)
//                         conv2 f=r/32,i=r%32 (f<8: nn_w, f==8: nn_b)
__global__ void k_wprep(const float* __restrict__ c1w, const float* __restrict__ c1b,
                        const float* __restrict__ c2w, const float* __restrict__ c2b,
                        __bf16* __restrict__ Wsw1, __bf16* __restrict__ Wsw2) {
  int t = blockIdx.x * blockDim.x + threadIdx.x;
  if (t < NSW1) {
    int j = t & 15, lane = (t >> 4) & 31, nt = (t >> 9) & 1, kb = t >> 10;
    int K = kb * 32 + ((lane >> 4) << 4) + j;
    int N = nt * 16 + (lane & 15);
    int f = K / 16, i = K % 16;
    float v = 0.0f;
    if (f < 8)       v = c1w[f * 512 + i * 32 + N];
    else if (f == 8) v = c1b[i * 32 + N];
    Wsw1[t] = (__bf16)v;
  } else if (t < NSW1 + NSW2) {
    int t2 = t - NSW1;
    int j = t2 & 15, lane = (t2 >> 4) & 31, nt = (t2 >> 9) & 1, kb = t2 >> 10;
    int K = kb * 32 + ((lane >> 4) << 4) + j;
    int N = nt * 16 + (lane & 15);
    int f = K / 32, i = K % 32;
    float v = (f < 8) ? c2w[f * 1024 + i * 32 + N] : c2b[i * 32 + N];
    Wsw2[t2] = (__bf16)v;
  }
}

// ---------------- fused NNConv edge GEMM (WMMA, bf16 in / f32 acc) ----------------
// msg[e,o] = sum_{f,i} ea'[e,f]*xin[src[e],i] * Wc[f*CIN+i, o] ; atomic scatter at dst.
// Each iteration: 32 edges = two 16-row M-tiles -> 4 independent accumulator chains.
template <int CIN, int KF>
__global__ __launch_bounds__(32) void k_conv_edges(
    const float* __restrict__ xin, const float* __restrict__ ea,
    const int* __restrict__ ei, const v16bf* __restrict__ Wf,
    float* __restrict__ agg) {
  constexpr int KBLK = (CIN * KF) / 32;
  __shared__ float xs[32][CIN];
  __shared__ float eas[32][KF];
  __shared__ int   srcs[32];
  __shared__ int   dsts[32];

  const int lane = threadIdx.x;
  const int lh   = lane >> 4;      // half-wave: 0 or 1
  const int M    = lane & 15;
  const int Ncol = lane & 15;

  for (int it = 0; it < 4; ++it) {
    const int ebase = blockIdx.x * 128 + it * 32;
    // stage 32 edges: one lane per edge (no divergence)
    {
      const int e = ebase + lane;
      srcs[lane] = ei[e];
      dsts[lane] = ei[EE + e];
#pragma unroll
      for (int k = 0; k < KF; ++k)
        eas[lane][k] = (k < 8) ? ea[e * 8 + k] : ((k == 8) ? 1.0f : 0.0f);
    }
    if (it < 3) {
      __builtin_prefetch(&ei[ebase + 32], 0, 0);
      __builtin_prefetch(&ea[(size_t)(ebase + 32) * 8], 0, 0);
    }
    __syncthreads();
    for (int idx = lane; idx < 32 * CIN; idx += 32) {
      const int m = idx / CIN, i = idx % CIN;
      xs[m][i] = xin[srcs[m] * CIN + i];
    }
    __syncthreads();

    v8f acc00 = {}, acc01 = {}, acc10 = {}, acc11 = {};
#pragma unroll
    for (int kb = 0; kb < KBLK; ++kb) {
      // A fragments for the two M-tiles (contiguous k order, same as B:
      // lane<16 holds K 0..15; lane>=16 holds K 16..31)
      v16bf a0, a1;
#pragma unroll
      for (int j = 0; j < 16; ++j) {
        const int kl = lh * 16 + j;
        const int kk = kb * 32 + kl;
        const int f = kk / CIN, i = kk % CIN;
        a0[j] = (__bf16)(eas[M][f] * xs[M][i]);
        a1[j] = (__bf16)(eas[16 + M][f] * xs[16 + M][i]);
      }
      // B fragments: one contiguous 32B vector load each (pre-swizzled)
      const v16bf b0 = Wf[(kb * 2 + 0) * 32 + lane];
      const v16bf b1 = Wf[(kb * 2 + 1) * 32 + lane];
      acc00 = __builtin_amdgcn_wmma_f32_16x16x32_bf16(false, a0, false, b0, (short)0, acc00, false, false);
      acc10 = __builtin_amdgcn_wmma_f32_16x16x32_bf16(false, a1, false, b0, (short)0, acc10, false, false);
      acc01 = __builtin_amdgcn_wmma_f32_16x16x32_bf16(false, a0, false, b1, (short)0, acc01, false, false);
      acc11 = __builtin_amdgcn_wmma_f32_16x16x32_bf16(false, a1, false, b1, (short)0, acc11, false, false);
    }

    // C/D layout: lane<16 -> N=lane, VGPR r -> M=r ; lane>=16 -> N=lane-16, M=r+8
    const int mb = lh * 8;
#pragma unroll
    for (int r = 0; r < 8; ++r) {
      const int d0 = dsts[mb + r];
      const int d1 = dsts[16 + mb + r];
      unsafeAtomicAdd(&agg[d0 * 32 + Ncol],       acc00[r]);
      unsafeAtomicAdd(&agg[d0 * 32 + 16 + Ncol],  acc01[r]);
      unsafeAtomicAdd(&agg[d1 * 32 + Ncol],       acc10[r]);
      unsafeAtomicAdd(&agg[d1 * 32 + 16 + Ncol],  acc11[r]);
    }
    __syncthreads();
  }
}

// ---------------- node update 1: h1 = leaky(agg + x@root + bias) ----------------
__global__ void k_node1(const float* __restrict__ agg, const float* __restrict__ x,
                        const float* __restrict__ root, const float* __restrict__ bias,
                        float* __restrict__ h1) {
  int t = blockIdx.x * blockDim.x + threadIdx.x;
  if (t >= NN * 32) return;
  int n = t / 32, o = t % 32;
  float acc = agg[t] + bias[o];
#pragma unroll
  for (int i = 0; i < 16; ++i) acc += x[n * 16 + i] * root[i * 32 + o];
  h1[t] = leaky(acc);
}

// ---------------- node update 2: h2 = leaky(agg + h1@root + bias) ----------------
__global__ void k_node2(const float* __restrict__ agg, const float* __restrict__ h1,
                        const float* __restrict__ root, const float* __restrict__ bias,
                        float* __restrict__ h2) {
  int t = blockIdx.x * blockDim.x + threadIdx.x;
  if (t >= NN * 32) return;
  int n = t / 32, o = t % 32;
  float acc = agg[t] + bias[o];
#pragma unroll
  for (int i = 0; i < 32; ++i) acc += h1[n * 32 + i] * root[i * 32 + o];
  h2[t] = leaky(acc);
}

// ---------------- gating: hfin = sigmoid(-clip(cat@sig_w+b)) * tanh(cat@tanh_w+b) ----------------
__global__ void k_gate(const float* __restrict__ h2, const float* __restrict__ x,
                       const float* __restrict__ sw, const float* __restrict__ sb,
                       const float* __restrict__ tw, const float* __restrict__ tb,
                       float* __restrict__ hfin) {
  int t = blockIdx.x * blockDim.x + threadIdx.x;
  if (t >= NN * 32) return;
  int n = t / 32, o = t % 32;
  float sx = sb[o], ty = tb[o];
#pragma unroll
  for (int c = 0; c < 48; ++c) {
    float cv = (c < 32) ? h2[n * 32 + c] : x[n * 16 + (c - 32)];
    sx += cv * sw[c * 32 + o];
    ty += cv * tw[c * 32 + o];
  }
  sx = fminf(fmaxf(sx, -30.0f), 30.0f);
  hfin[t] = (1.0f / (1.0f + expf(sx))) * tanhf(ty);
}

// ---------------- per-graph pooling + head ----------------
__global__ __launch_bounds__(256) void k_final(
    const float* __restrict__ x, const float* __restrict__ ea,
    const float* __restrict__ hfin,
    const float* __restrict__ lx0w, const float* __restrict__ lx0b,
    const float* __restrict__ le0w, const float* __restrict__ le0b,
    const float* __restrict__ le1w, const float* __restrict__ le1b,
    const float* __restrict__ f1w, const float* __restrict__ f1b,
    const float* __restrict__ f2w, const float* __restrict__ f2b,
    float* __restrict__ out) {
  __shared__ float x0s[16];
  __shared__ float es[8];
  __shared__ float ox[32];
  __shared__ float sp[64];
  __shared__ float red[256];
  const int g = blockIdx.x, t = threadIdx.x;
  if (t < 16) x0s[t] = 0.0f;
  if (t < 8)  es[t]  = 0.0f;
  if (t < 32) ox[t]  = 0.0f;
  __syncthreads();
  for (int n = t; n < NPG; n += 256) {
    const int gn = g * NPG + n;
#pragma unroll
    for (int i = 0; i < 16; ++i) atomicAdd(&x0s[i], x[gn * 16 + i]);
#pragma unroll
    for (int o = 0; o < 32; ++o) atomicAdd(&ox[o], hfin[gn * 32 + o]);
  }
  for (int e = t; e < EPG; e += 256) {
    const int ge = g * EPG + e;
#pragma unroll
    for (int i = 0; i < 8; ++i) atomicAdd(&es[i], ea[ge * 8 + i]);
  }
  __syncthreads();
  if (t < 16) {
    float acc = lx0b[t];
#pragma unroll
    for (int i = 0; i < 16; ++i) acc += (x0s[i] / (float)NPG) * lx0w[i * 16 + t];
    sp[t] = acc;
  } else if (t < 24) {
    const int c = t - 16;
    float acc = le0b[c];
#pragma unroll
    for (int i = 0; i < 8; ++i) acc += (es[i] * 0.05f) * le0w[i * 8 + c];
    sp[t] = acc;
  } else if (t < 32) {
    const int c = t - 24;
    float acc = le1b[c];
#pragma unroll
    for (int i = 0; i < 8; ++i) acc += (es[i] / (float)EPG) * le1w[i * 8 + c];
    sp[t] = acc;
  } else if (t < 64) {
    sp[t] = ox[t - 32];
  }
  __syncthreads();
  float acc = f1b[t];
#pragma unroll
  for (int c = 0; c < 64; ++c) acc += sp[c] * f1w[c * 256 + t];
  red[t] = leaky(acc) * f2w[t];
  __syncthreads();
  for (int s = 128; s > 0; s >>= 1) {
    if (t < s) red[t] += red[t + s];
    __syncthreads();
  }
  if (t == 0) out[g] = tanhf((red[0] + f2b[0]) * 0.25f);
}

extern "C" void kernel_launch(void* const* d_in, const int* in_sizes, int n_in,
                              void* d_out, int out_size, void* d_ws, size_t ws_size,
                              hipStream_t stream) {
  const float* x       = (const float*)d_in[0];
  const float* ea      = (const float*)d_in[1];
  const int*   ei      = (const int*)d_in[2];
  const float* lx0w    = (const float*)d_in[5];
  const float* lx0b    = (const float*)d_in[6];
  const float* le0w    = (const float*)d_in[7];
  const float* le0b    = (const float*)d_in[8];
  const float* le1w    = (const float*)d_in[9];
  const float* le1b    = (const float*)d_in[10];
  const float* c1w     = (const float*)d_in[11];
  const float* c1b     = (const float*)d_in[12];
  const float* c1root  = (const float*)d_in[13];
  const float* c1bias  = (const float*)d_in[14];
  const float* c2w     = (const float*)d_in[15];
  const float* c2b     = (const float*)d_in[16];
  const float* c2root  = (const float*)d_in[17];
  const float* c2bias  = (const float*)d_in[18];
  const float* sw      = (const float*)d_in[19];
  const float* sb      = (const float*)d_in[20];
  const float* tw      = (const float*)d_in[21];
  const float* tb      = (const float*)d_in[22];
  const float* f1w     = (const float*)d_in[23];
  const float* f1b     = (const float*)d_in[24];
  const float* f2w     = (const float*)d_in[25];
  const float* f2b     = (const float*)d_in[26];

  char* ws = (char*)d_ws;
  const size_t NB = (size_t)NN * 32 * sizeof(float);  // 1,280,000 bytes
  float*  h1agg = (float*)(ws + 0 * NB);
  float*  h1    = (float*)(ws + 1 * NB);
  float*  h2agg = (float*)(ws + 2 * NB);
  float*  h2    = (float*)(ws + 3 * NB);
  float*  hfin  = (float*)(ws + 4 * NB);
  __bf16* Wsw1  = (__bf16*)(ws + 5 * NB);
  __bf16* Wsw2  = (__bf16*)(ws + 5 * NB + (size_t)NSW1 * sizeof(__bf16));

  float* out = (float*)d_out;

  k_init<<<640, 256, 0, stream>>>(h1agg, h2agg, NN * 32);
  k_wprep<<<(NSW1 + NSW2 + 255) / 256, 256, 0, stream>>>(c1w, c1b, c2w, c2b, Wsw1, Wsw2);

  // conv1: K = 10*16 = 160 (5 WMMA K-blocks), conv2: K = 9*32 = 288 (9 K-blocks)
  k_conv_edges<16, 10><<<EE / 128, 32, 0, stream>>>(x, ea, ei, (const v16bf*)Wsw1, h1agg);
  k_node1<<<(NN * 32 + 255) / 256, 256, 0, stream>>>(h1agg, x, c1root, c1bias, h1);
  k_conv_edges<32, 9><<<EE / 128, 32, 0, stream>>>(h1, ea, ei, (const v16bf*)Wsw2, h2agg);
  k_node2<<<(NN * 32 + 255) / 256, 256, 0, stream>>>(h2agg, h1, c2root, c2bias, h2);
  k_gate<<<(NN * 32 + 255) / 256, 256, 0, stream>>>(h2, x, sw, sb, tw, tb, hfin);
  k_final<<<GG, 256, 0, stream>>>(x, ea, hfin, lx0w, lx0b, le0w, le0b, le1w, le1b,
                                  f1w, f1b, f2w, f2b, out);
}
